// Decoder_3667902071347
// MI455X (gfx1250) — matrix operand order repaired and checked by reading, not compiled, not run on hardware
//
#include <hip/hip_runtime.h>
#include <hip/hip_bf16.h>
#include <math.h>

// ---------------------------------------------------------------------------
// Attention-GRU decoder for MI455X (gfx1250), wave32 + WMMA bf16.
// H=512 E=256 S=32 B=1024 T=32 V=256. One block = 16 batch rows, loops over
// all T steps with h0/h1 resident in LDS; all GEMMs via v_wmma_f32_16x16x32_bf16.
// Transcendentals use gfx1250 hardware TRANS ops (v_tanh_f32 / v_exp_f32).
// ---------------------------------------------------------------------------

typedef __bf16 bf16_t;
typedef __attribute__((ext_vector_type(16))) __bf16 v16bf;
typedef __attribute__((ext_vector_type(8)))  __bf16 v8bf;
typedef __attribute__((ext_vector_type(8)))  float  v8f;

#define HD 512
#define ED 256
#define SD 32
#define BD 1024
#define TD 32
#define VD 256
#define START_TOK 1

// padded LDS strides (bf16 elems): keep 16B alignment, dodge bank conflicts
#define HP 520
#define GP 776

#define WMMA_BF16(acc, a, b)                                                   \
  acc = __builtin_amdgcn_wmma_f32_16x16x32_bf16(false, (a), false, (b),        \
                                                (short)0, (acc), false, false)

static __device__ __forceinline__ float fast_tanh(float x) {
#if __has_builtin(__builtin_amdgcn_tanhf)
  return __builtin_amdgcn_tanhf(x);   // gfx1250 v_tanh_f32 (TRANS, coexec w/ WMMA)
#else
  return tanhf(x);
#endif
}
static __device__ __forceinline__ float fast_exp(float x) { return __expf(x); }
static __device__ __forceinline__ float fast_log(float x) { return __logf(x); }
static __device__ __forceinline__ float fast_sigmoid(float x) {
  return 1.0f / (1.0f + fast_exp(-x));
}

// B fragment: packed fragment-major layout, one contiguous 32B load per lane.
static __device__ __forceinline__ v16bf load_b_frag(const bf16_t* pk, int kt,
                                                    int nt, int ntn, int lane) {
  const bf16_t* p = pk + ((((size_t)kt * ntn + nt) * 32 + lane) << 4);
  return *(const v16bf*)p;
}

// A fragment from LDS bf16 row-major (stride in elems, mult of 8).
// 16-bit A 16x32 layout: lane m=l&15, K(e) = k0 + 8*(l>>4) + (e<8 ? e : e+8).
static __device__ __forceinline__ v16bf load_a_lds(const bf16_t* base,
                                                   int stride, int k0, int lane) {
  int m  = lane & 15;
  int hi = (lane >> 4) * 8;
  const v8bf* p0 = (const v8bf*)(base + m * stride + k0 + hi);
  const v8bf* p1 = (const v8bf*)(base + m * stride + k0 + 16 + hi);
  v8bf lo = *p0, hv = *p1;
  return __builtin_shufflevector(lo, hv, 0, 1, 2, 3, 4, 5, 6, 7,
                                 8, 9, 10, 11, 12, 13, 14, 15);
}

// A fragment straight from global fp32 (row-major, ld elems), converted inline.
static __device__ __forceinline__ v16bf load_a_gf32(const float* A, size_t ld,
                                                    int mbase, int k0, int lane) {
  int m  = lane & 15;
  int hi = (lane >> 4) * 8;
  const float* p0 = A + (size_t)(mbase + m) * ld + k0 + hi;
  const float* p1 = p0 + 16;
  v16bf a;
#pragma unroll
  for (int i = 0; i < 8; ++i) a[i] = (bf16_t)p0[i];
#pragma unroll
  for (int i = 0; i < 8; ++i) a[8 + i] = (bf16_t)p1[i];
  return a;
}

// ---------------------------------------------------------------------------
// Pack a (transposed / gate-sliced / K-concatenated) weight into fragment-major
// bf16. Logical B element (k,n) = src1[nOff1+n][k] for k<K1 else
// src2[nOff2+n][k-K1]. dst frag index = ((kt*ntn+nt)*32+lane)*16+e with
// k = kt*32 + 8*(lane>>4) + (e<8 ? e : e+8), n = nt*16 + (lane&15).
// ---------------------------------------------------------------------------
__global__ void pack_b_kernel(bf16_t* __restrict__ dst,
                              const float* __restrict__ src1, int K1, int nOff1,
                              const float* __restrict__ src2, int K2, int nOff2,
                              int Ktot, int N) {
  int ntn   = N >> 4;
  int total = (Ktot >> 5) * ntn * 32;
  int tid   = blockIdx.x * blockDim.x + threadIdx.x;
  if (tid >= total) return;
  int lane = tid & 31;
  int nt   = (tid >> 5) % ntn;
  int kt   = (tid >> 5) / ntn;
  int n    = nt * 16 + (lane & 15);
  int kb   = kt * 32 + ((lane >> 4) * 8);
  bf16_t* out = dst + ((size_t)tid << 4);
#pragma unroll
  for (int e = 0; e < 16; ++e) {
    int   k = kb + (e < 8 ? e : e + 8);
    float v = (k < K1) ? src1[(size_t)(nOff1 + n) * K1 + k]
                       : src2[(size_t)(nOff2 + n) * K2 + (k - K1)];
    out[e] = (bf16_t)v;
  }
}

// ---------------------------------------------------------------------------
// k_proj = enc(S*B x H) @ Ua^T + bUa  (fp32 out).  grid.x = S*B/16 blocks,
// 8 waves each cover 64 columns.
// ---------------------------------------------------------------------------
__global__ __launch_bounds__(256) void kproj_kernel(
    const float* __restrict__ enc, const bf16_t* __restrict__ UaPk,
    const float* __restrict__ bUa, float* __restrict__ kproj) {
  int lane  = threadIdx.x & 31;
  int wave  = threadIdx.x >> 5;
  int mbase = blockIdx.x * 16;
  int n0    = wave * 64;
  v8f acc[4] = {};
  for (int kt = 0; kt < HD / 32; ++kt) {
    v16bf a = load_a_gf32(enc, HD, mbase, kt * 32, lane);
#pragma unroll
    for (int j = 0; j < 4; ++j) {
      v16bf b = load_b_frag(UaPk, kt, (n0 >> 4) + j, HD >> 4, lane);
      WMMA_BF16(acc[j], a, b);
    }
  }
  int mhi = 8 * (lane >> 4), nl = lane & 15;
#pragma unroll
  for (int j = 0; j < 4; ++j) {
    int   col  = n0 + j * 16 + nl;
    float bias = bUa[col];
#pragma unroll
    for (int i = 0; i < 8; ++i)
      kproj[(size_t)(mbase + mhi + i) * HD + col] = acc[j][i] + bias;
  }
}

// ---------------------------------------------------------------------------
// Persistent decode kernel: 64 blocks x 256 threads, T-step loop inside.
// ---------------------------------------------------------------------------
__global__ __launch_bounds__(256) void decode_kernel(
    const float* __restrict__ dstate, const float* __restrict__ enc,
    const int* __restrict__ tgt, const float* __restrict__ emb,
    const float* __restrict__ bWa, const float* __restrict__ Va,
    const float* __restrict__ bVa, const float* __restrict__ b_ih0,
    const float* __restrict__ b_hh0, const float* __restrict__ b_ih1,
    const float* __restrict__ b_hh1, const float* __restrict__ fcb,
    const bf16_t* __restrict__ WaPk, const bf16_t* __restrict__ Wc0Pk,
    const bf16_t* __restrict__ Wi0nPk, const bf16_t* __restrict__ Wh0nPk,
    const bf16_t* __restrict__ Wc1Pk, const bf16_t* __restrict__ Wi1nPk,
    const bf16_t* __restrict__ Wh1nPk, const bf16_t* __restrict__ fcPk,
    const float* __restrict__ kproj, float* __restrict__ out_dec,
    float* __restrict__ out_attn, float* __restrict__ part_loss,
    float* __restrict__ part_corr) {
  // scr serves (phase-disjoint): q_proj [16][516] f32 | rz gates [16][1028] |
  // logits [16][260]
  __shared__ __align__(16) float  scr[16 * 1028];
  __shared__ __align__(16) bf16_t sh_h0[16 * HP];
  __shared__ __align__(16) bf16_t sh_h1[16 * HP];
  __shared__ __align__(16) bf16_t sh_gin[16 * GP];   // [x(256) | context(512)]
  __shared__ float sh_e[SD * 16];    // scores; reused as reduction scratch
  __shared__ float sh_w[SD * 16];    // attention weights
  __shared__ int   sh_idx[256];      // argmax reduction scratch
  __shared__ int   sh_tok[16];
  __shared__ float sh_red[32];       // per-b max / final partials

  const int tid = threadIdx.x, lane = tid & 31, wave = tid >> 5;
  const int bt0 = blockIdx.x * 16;
  const int mhi = 8 * (lane >> 4), nl = lane & 15;

  // init recurrent state + start tokens
  for (int i = tid; i < 16 * HD; i += 256) {
    int b = i >> 9, h = i & (HD - 1);
    sh_h0[b * HP + h] = (bf16_t)dstate[(size_t)(bt0 + b) * HD + h];
    sh_h1[b * HP + h] = (bf16_t)dstate[(size_t)BD * HD + (size_t)(bt0 + b) * HD + h];
  }
  if (tid < 16) sh_tok[tid] = START_TOK;
  float loss_acc = 0.0f;
  int   corr_flag = 1;
  __syncthreads();

  for (int t = 0; t < TD; ++t) {
    // ---- phase 0: x = relu(emb[tok]) -> gin[:,0:256] (bf16) -------------
    {
      int b = tid >> 4, e0 = (tid & 15) * 16;
      const float* ep = emb + (size_t)sh_tok[b] * ED + e0;
      bf16_t*      gp = sh_gin + b * GP + e0;
#pragma unroll
      for (int j = 0; j < 16; ++j) gp[j] = (bf16_t)fmaxf(ep[j], 0.0f);
    }

    // ---- phase 1: q_proj = h1 @ Wa^T + bWa -> scr[b*516+h] --------------
    {
      v8f acc[4] = {};
      int n0 = wave * 64;
      for (int kt = 0; kt < 16; ++kt) {
        v16bf a = load_a_lds(sh_h1, HP, kt * 32, lane);
#pragma unroll
        for (int j = 0; j < 4; ++j)
          WMMA_BF16(acc[j], a, load_b_frag(WaPk, kt, (n0 >> 4) + j, 32, lane));
      }
#pragma unroll
      for (int j = 0; j < 4; ++j) {
        int   col  = n0 + j * 16 + nl;
        float bias = bWa[col];
#pragma unroll
        for (int i = 0; i < 8; ++i)
          scr[(mhi + i) * 516 + col] = acc[j][i] + bias;
      }
    }
    __syncthreads();

    // ---- phase 2: e[s,b] = Va . tanh(q[b] + kproj[s,b]) + bVa -----------
    for (int p = tid; p < SD * 16; p += 256) {
      int s = p >> 4, b = p & 15;
      const float4* kp4 = (const float4*)(kproj + ((size_t)s * BD + bt0 + b) * HD);
      const float4* qp4 = (const float4*)(scr + b * 516);
      float acc = 0.0f;
#pragma unroll 4
      for (int h4 = 0; h4 < HD / 4; ++h4) {
        float4       k  = kp4[h4];
        float4       q  = qp4[h4];
        const float* vp = Va + h4 * 4;  // wave-uniform scalar loads
        acc += vp[0] * fast_tanh(q.x + k.x);
        acc += vp[1] * fast_tanh(q.y + k.y);
        acc += vp[2] * fast_tanh(q.z + k.z);
        acc += vp[3] * fast_tanh(q.w + k.w);
      }
      sh_e[s * 16 + b] = acc + bVa[0];
    }
    __syncthreads();

    // ---- phase 3: softmax over s; emit attention weights ----------------
    if (tid < 16) {
      int   b = tid;
      float m = -1e30f;
      for (int s = 0; s < SD; ++s) m = fmaxf(m, sh_e[s * 16 + b]);
      float sum = 0.0f;
      for (int s = 0; s < SD; ++s) {
        float w = fast_exp(sh_e[s * 16 + b] - m);
        sh_w[s * 16 + b] = w;
        sum += w;
      }
      float inv = 1.0f / sum;
      for (int s = 0; s < SD; ++s) {
        float w = sh_w[s * 16 + b] * inv;
        sh_w[s * 16 + b] = w;
        out_attn[((size_t)t * BD + bt0 + b) * SD + s] = w;  // (T,B,1,S)
      }
    }
    __syncthreads();

    // ---- phase 4: context[b,h] = sum_s w*enc -> gin[:,256:768] ----------
    {
      int   b = tid >> 4, hb = (tid & 15) * 32;
      float ctx[32] = {};
      for (int s = 0; s < SD; ++s) {
        float        w  = sh_w[s * 16 + b];
        const float* ep = enc + ((size_t)s * BD + bt0 + b) * HD + hb;
#pragma unroll
        for (int j = 0; j < 32; ++j) ctx[j] += w * ep[j];
      }
      bf16_t* gp = sh_gin + b * GP + ED + hb;
#pragma unroll
      for (int j = 0; j < 32; ++j) gp[j] = (bf16_t)ctx[j];
    }
    __syncthreads();

    // ---- phase 5: GRU0 r,z = sigmoid([gin|h0] @ Wc0 + b) -> scr ---------
    {
      v8f acc[8] = {};
      int n0 = wave * 128;
      for (int kt = 0; kt < 40; ++kt) {
        v16bf a = (kt < 24) ? load_a_lds(sh_gin, GP, kt * 32, lane)
                            : load_a_lds(sh_h0, HP, (kt - 24) * 32, lane);
#pragma unroll
        for (int j = 0; j < 8; ++j)
          WMMA_BF16(acc[j], a, load_b_frag(Wc0Pk, kt, (n0 >> 4) + j, 64, lane));
      }
#pragma unroll
      for (int j = 0; j < 8; ++j) {
        int   col  = n0 + j * 16 + nl;
        float bias = b_ih0[col] + b_hh0[col];
#pragma unroll
        for (int i = 0; i < 8; ++i)
          scr[(mhi + i) * 1028 + col] = fast_sigmoid(acc[j][i] + bias);
      }
    }
    __syncthreads();

    // ---- phase 6: GRU0 n-gate + h0 update -------------------------------
    {
      v8f ai[4] = {}, ah[4] = {};
      int n0 = wave * 64;
      for (int kt = 0; kt < 24; ++kt) {
        v16bf a = load_a_lds(sh_gin, GP, kt * 32, lane);
#pragma unroll
        for (int j = 0; j < 4; ++j)
          WMMA_BF16(ai[j], a, load_b_frag(Wi0nPk, kt, (n0 >> 4) + j, 32, lane));
      }
      for (int kt = 0; kt < 16; ++kt) {
        v16bf a = load_a_lds(sh_h0, HP, kt * 32, lane);
#pragma unroll
        for (int j = 0; j < 4; ++j)
          WMMA_BF16(ah[j], a, load_b_frag(Wh0nPk, kt, (n0 >> 4) + j, 32, lane));
      }
      __syncthreads();  // all sh_h0 reads done before overwrite
#pragma unroll
      for (int j = 0; j < 4; ++j) {
        int   col = n0 + j * 16 + nl;
        float bi = b_ih0[1024 + col], bh = b_hh0[1024 + col];
#pragma unroll
        for (int i = 0; i < 8; ++i) {
          int   m  = mhi + i;
          float r  = scr[m * 1028 + col];
          float z  = scr[m * 1028 + 512 + col];
          float nn = fast_tanh(ai[j][i] + bi + r * (ah[j][i] + bh));
          float hp = (float)sh_h0[m * HP + col];
          sh_h0[m * HP + col] = (bf16_t)((1.0f - z) * nn + z * hp);
        }
      }
    }
    __syncthreads();

    // ---- phase 7: GRU1 r,z = sigmoid([h0|h1] @ Wc1 + b) -> scr ----------
    {
      v8f acc[8] = {};
      int n0 = wave * 128;
      for (int kt = 0; kt < 32; ++kt) {
        v16bf a = (kt < 16) ? load_a_lds(sh_h0, HP, kt * 32, lane)
                            : load_a_lds(sh_h1, HP, (kt - 16) * 32, lane);
#pragma unroll
        for (int j = 0; j < 8; ++j)
          WMMA_BF16(acc[j], a, load_b_frag(Wc1Pk, kt, (n0 >> 4) + j, 64, lane));
      }
#pragma unroll
      for (int j = 0; j < 8; ++j) {
        int   col  = n0 + j * 16 + nl;
        float bias = b_ih1[col] + b_hh1[col];
#pragma unroll
        for (int i = 0; i < 8; ++i)
          scr[(mhi + i) * 1028 + col] = fast_sigmoid(acc[j][i] + bias);
      }
    }
    __syncthreads();

    // ---- phase 8: GRU1 n-gate + h1 update -------------------------------
    {
      v8f ai[4] = {}, ah[4] = {};
      int n0 = wave * 64;
      for (int kt = 0; kt < 16; ++kt) {
        v16bf a = load_a_lds(sh_h0, HP, kt * 32, lane);
#pragma unroll
        for (int j = 0; j < 4; ++j)
          WMMA_BF16(ai[j], a, load_b_frag(Wi1nPk, kt, (n0 >> 4) + j, 32, lane));
      }
      for (int kt = 0; kt < 16; ++kt) {
        v16bf a = load_a_lds(sh_h1, HP, kt * 32, lane);
#pragma unroll
        for (int j = 0; j < 4; ++j)
          WMMA_BF16(ah[j], a, load_b_frag(Wh1nPk, kt, (n0 >> 4) + j, 32, lane));
      }
      __syncthreads();
#pragma unroll
      for (int j = 0; j < 4; ++j) {
        int   col = n0 + j * 16 + nl;
        float bi = b_ih1[1024 + col], bh = b_hh1[1024 + col];
#pragma unroll
        for (int i = 0; i < 8; ++i) {
          int   m  = mhi + i;
          float r  = scr[m * 1028 + col];
          float z  = scr[m * 1028 + 512 + col];
          float nn = fast_tanh(ai[j][i] + bi + r * (ah[j][i] + bh));
          float hp = (float)sh_h1[m * HP + col];
          sh_h1[m * HP + col] = (bf16_t)((1.0f - z) * nn + z * hp);
        }
      }
    }
    __syncthreads();

    // ---- phase 9: logits = h1 @ fcW^T + fcb -> scr[b*260+v] -------------
    {
      v8f acc[2] = {};
      int n0 = wave * 32;
      for (int kt = 0; kt < 16; ++kt) {
        v16bf a = load_a_lds(sh_h1, HP, kt * 32, lane);
#pragma unroll
        for (int j = 0; j < 2; ++j)
          WMMA_BF16(acc[j], a, load_b_frag(fcPk, kt, (n0 >> 4) + j, 16, lane));
      }
#pragma unroll
      for (int j = 0; j < 2; ++j) {
        int   col  = n0 + j * 16 + nl;
        float bias = fcb[col];
#pragma unroll
        for (int i = 0; i < 8; ++i)
          scr[(mhi + i) * 260 + col] = acc[j][i] + bias;
      }
    }
    __syncthreads();

    // ---- phase 10: log_softmax/argmax/NLL, parallel across all threads --
    // stage A: per-thread chunk max/argmax (b = tid>>4, 16 contiguous logits)
    {
      int          b  = tid >> 4, c16 = tid & 15;
      const float* lg = scr + b * 260 + c16 * 16;
      float        mx = lg[0];
      int          am = c16 * 16;
#pragma unroll
      for (int j = 1; j < 16; ++j)
        if (lg[j] > mx) { mx = lg[j]; am = c16 * 16 + j; }
      sh_e[tid]   = mx;
      sh_idx[tid] = am;
    }
    __syncthreads();
    // stage B: 16 threads finalize max/argmax (first-index tie-break)
    if (tid < 16) {
      int   b  = tid;
      float mx = sh_e[b * 16];
      int   am = sh_idx[b * 16];
      for (int j = 1; j < 16; ++j) {
        float v = sh_e[b * 16 + j];
        if (v > mx) { mx = v; am = sh_idx[b * 16 + j]; }
      }
      sh_red[b]  = mx;
      sh_tok[b]  = am;  // next token (greedy); consumed next iteration only
    }
    __syncthreads();
    // stage C: per-thread chunk exp-sum with global max
    {
      int          b  = tid >> 4, c16 = tid & 15;
      const float* lg = scr + b * 260 + c16 * 16;
      float        mx = sh_red[b];
      float        s  = 0.0f;
#pragma unroll
      for (int j = 0; j < 16; ++j) s += fast_exp(lg[j] - mx);
      sh_e[tid] = s;
    }
    __syncthreads();
    // stage D: 16 threads finalize lse, loss, decoded output
    if (tid < 16) {
      int   b   = tid;
      float sum = 0.0f;
      for (int j = 0; j < 16; ++j) sum += sh_e[b * 16 + j];
      float lse = sh_red[b] + fast_log(sum);
      int   tg  = tgt[(size_t)(bt0 + b) * TD + t];
      loss_acc += -(scr[b * 260 + tg] - lse) * (1.0f / (float)BD);
      int am = sh_tok[b];
      out_dec[(size_t)(bt0 + b) * TD + t] = (float)am;
      if (am != tg) corr_flag = 0;
    }
    __syncthreads();
  }

  // deterministic per-block partials
  if (tid < 16) {
    sh_red[tid]      = loss_acc;
    sh_red[16 + tid] = (float)corr_flag;
  }
  __syncthreads();
  if (tid == 0) {
    float ls = 0.0f, cs = 0.0f;
    for (int i = 0; i < 16; ++i) { ls += sh_red[i]; cs += sh_red[16 + i]; }
    part_loss[blockIdx.x] = ls;
    part_corr[blockIdx.x] = cs;
  }
}

__global__ void reduce_kernel(const float* __restrict__ part_loss,
                              const float* __restrict__ part_corr,
                              float* __restrict__ out_loss,
                              float* __restrict__ out_corr) {
  if (blockIdx.x == 0 && threadIdx.x == 0) {
    float l = 0.0f, c = 0.0f;
    for (int i = 0; i < BD / 16; ++i) { l += part_loss[i]; c += part_corr[i]; }
    out_loss[0] = l;
    out_corr[0] = c;
  }
}

// ---------------------------------------------------------------------------
extern "C" void kernel_launch(void* const* d_in, const int* in_sizes, int n_in,
                              void* d_out, int out_size, void* d_ws,
                              size_t ws_size, hipStream_t stream) {
  (void)in_sizes; (void)n_in; (void)out_size; (void)ws_size;
  const float* dstate = (const float*)d_in[0];
  const float* enc    = (const float*)d_in[1];
  const int*   tgtb   = (const int*)d_in[2];
  const float* emb    = (const float*)d_in[3];
  const float* Wa     = (const float*)d_in[4];
  const float* bWa    = (const float*)d_in[5];
  const float* Ua     = (const float*)d_in[6];
  const float* bUa    = (const float*)d_in[7];
  const float* Va     = (const float*)d_in[8];
  const float* bVa    = (const float*)d_in[9];
  const float* W_ih0  = (const float*)d_in[10];
  const float* W_hh0  = (const float*)d_in[11];
  const float* b_ih0  = (const float*)d_in[12];
  const float* b_hh0  = (const float*)d_in[13];
  const float* W_ih1  = (const float*)d_in[14];
  const float* W_hh1  = (const float*)d_in[15];
  const float* b_ih1  = (const float*)d_in[16];
  const float* b_hh1  = (const float*)d_in[17];
  const float* fcW    = (const float*)d_in[18];
  const float* fcb    = (const float*)d_in[19];

  // workspace carve-up (256B aligned)
  unsigned char* ws  = (unsigned char*)d_ws;
  size_t         off = 0;
  auto carve = [&](size_t bytes) {
    size_t o = off;
    off = (off + bytes + 255) & ~(size_t)255;
    return (void*)(ws + o);
  };
  float*  kproj  = (float*)carve((size_t)SD * BD * HD * sizeof(float));
  bf16_t* UaPk   = (bf16_t*)carve((size_t)512 * 512 * 2);
  bf16_t* WaPk   = (bf16_t*)carve((size_t)512 * 512 * 2);
  bf16_t* Wc0Pk  = (bf16_t*)carve((size_t)1280 * 1024 * 2);
  bf16_t* Wi0nPk = (bf16_t*)carve((size_t)768 * 512 * 2);
  bf16_t* Wh0nPk = (bf16_t*)carve((size_t)512 * 512 * 2);
  bf16_t* Wc1Pk  = (bf16_t*)carve((size_t)1024 * 1024 * 2);
  bf16_t* Wi1nPk = (bf16_t*)carve((size_t)512 * 512 * 2);
  bf16_t* Wh1nPk = (bf16_t*)carve((size_t)512 * 512 * 2);
  bf16_t* fcPk   = (bf16_t*)carve((size_t)512 * 256 * 2);
  float*  pLoss  = (float*)carve(64 * sizeof(float));
  float*  pCorr  = (float*)carve(64 * sizeof(float));

  auto pack = [&](bf16_t* dst, const float* s1, int K1, int nOff1,
                  const float* s2, int K2, int nOff2, int Ktot, int N) {
    int total = (Ktot / 32) * (N / 16) * 32;
    pack_b_kernel<<<(total + 255) / 256, 256, 0, stream>>>(dst, s1, K1, nOff1,
                                                           s2, K2, nOff2, Ktot, N);
  };
  // transposed weights, gate-sliced / K-concatenated where useful
  pack(UaPk, Ua, 512, 0, Ua, 512, 0, 512, 512);
  pack(WaPk, Wa, 512, 0, Wa, 512, 0, 512, 512);
  pack(Wc0Pk, W_ih0, 768, 0, W_hh0, 512, 0, 1280, 1024);   // [gin|h0] -> r,z
  pack(Wi0nPk, W_ih0, 768, 1024, W_ih0, 768, 1024, 768, 512);
  pack(Wh0nPk, W_hh0, 512, 1024, W_hh0, 512, 1024, 512, 512);
  pack(Wc1Pk, W_ih1, 512, 0, W_hh1, 512, 0, 1024, 1024);   // [h0|h1] -> r,z
  pack(Wi1nPk, W_ih1, 512, 1024, W_ih1, 512, 1024, 512, 512);
  pack(Wh1nPk, W_hh1, 512, 1024, W_hh1, 512, 1024, 512, 512);
  pack(fcPk, fcW, 512, 0, fcW, 512, 0, 512, 256);

  // k_proj precompute (hoisted out of the scan, like the reference)
  kproj_kernel<<<(SD * BD) / 16, 256, 0, stream>>>(enc, UaPk, bUa, kproj);

  // outputs: decoded (B*T) | attentions (T*B*1*S) | loss | correct
  float* out_dec  = (float*)d_out;
  float* out_attn = out_dec + (size_t)BD * TD;
  float* out_loss = out_attn + (size_t)TD * BD * SD;
  float* out_corr = out_loss + 1;

  decode_kernel<<<BD / 16, 256, 0, stream>>>(
      dstate, enc, tgtb, emb, bWa, Va, bVa, b_ih0, b_hh0, b_ih1, b_hh1, fcb,
      WaPk, Wc0Pk, Wi0nPk, Wh0nPk, Wc1Pk, Wi1nPk, Wh1nPk, fcPk, kproj,
      out_dec, out_attn, pLoss, pCorr);

  reduce_kernel<<<1, 32, 0, stream>>>(pLoss, pCorr, out_loss, out_corr);
}
